// SwinTransformerBlock_64896955843044
// MI455X (gfx1250) — matrix operand order repaired and checked by reading, compiled, not run on hardware
//
#include <hip/hip_runtime.h>
#include <hip/hip_bf16.h>
#include <math.h>
#include <stdint.h>

typedef unsigned short u16;
typedef unsigned int   u32;
typedef __attribute__((ext_vector_type(16))) __bf16 v16bf;
typedef __attribute__((ext_vector_type(8)))  float  v8f;

union FragU { u32 u[8]; v16bf v; };

__device__ __forceinline__ u16 f2bf(float f) {
  u32 u = __float_as_uint(f);
  u32 r = u + 0x7FFFu + ((u >> 16) & 1u);   // round-to-nearest-even
  return (u16)(r >> 16);
}
__device__ __forceinline__ float gelu_exact(float x) {
  return 0.5f * x * (1.0f + erff(x * 0.70710678118654752f));
}
// gfx1250 async copy: global -> LDS, 16 bytes per lane, tracked by ASYNCcnt
__device__ __forceinline__ void async_ld16(u32 lds, u32 goff, const u16* base) {
  asm volatile("global_load_async_to_lds_b128 %0, %1, %2"
               :: "v"(lds), "v"(goff), "s"(base) : "memory");
}

// ---------------------------------------------------------------- weights cvt
// dst is bf16 TRANSPOSED [N][K] so GEMM B-tiles are contiguous row copies.
__global__ void cvt_bf16_t_kernel(const float* __restrict__ s, u16* __restrict__ d,
                                  int K, int N) {
  int i = blockIdx.x * 256 + threadIdx.x;
  if (i < K * N) {
    int k = i % K, n = i / K;
    d[i] = f2bf(s[(size_t)k * N + n]);
  }
}

// ------------------------------------------------- LayerNorm (+shift/window)
// MODE 0: dst row r is (window,token) order; source gathered with roll(-3,-3)
// MODE 1: identity row mapping (LN2)
template<int MODE>
__global__ __launch_bounds__(128)
void ln_kernel(const float* __restrict__ x, const float* __restrict__ g,
               const float* __restrict__ bta, u16* __restrict__ dst) {
  int r = blockIdx.x;                 // 0..100351
  int srow;
  if (MODE == 0) {
    int b = r / 3136, rem = r % 3136;
    int wi = rem / 49, t = rem % 49;
    int h = (wi >> 3) * 7 + t / 7;
    int w = (wi & 7) * 7 + t % 7;
    int hs = h + 3; if (hs >= 56) hs -= 56;
    int ws2 = w + 3; if (ws2 >= 56) ws2 -= 56;
    srow = b * 3136 + hs * 56 + ws2;
  } else {
    srow = r;
  }
  const float* xr = x + (size_t)srow * 384;
  int tid = threadIdx.x;              // 128 threads, 3 elems each
  float v0 = xr[tid], v1 = xr[tid + 128], v2 = xr[tid + 256];
  __shared__ float rs[128], rq[128];
  rs[tid] = v0 + v1 + v2;
  rq[tid] = v0 * v0 + v1 * v1 + v2 * v2;
  __syncthreads();
  for (int o = 64; o > 0; o >>= 1) {
    if (tid < o) { rs[tid] += rs[tid + o]; rq[tid] += rq[tid + o]; }
    __syncthreads();
  }
  float mean = rs[0] * (1.0f / 384.0f);
  float var  = rq[0] * (1.0f / 384.0f) - mean * mean;
  float inv  = rsqrtf(var + 1e-5f);
  u16* dr = dst + (size_t)r * 384;
  dr[tid]       = f2bf((v0 - mean) * inv * g[tid]       + bta[tid]);
  dr[tid + 128] = f2bf((v1 - mean) * inv * g[tid + 128] + bta[tid + 128]);
  dr[tid + 256] = f2bf((v2 - mean) * inv * g[tid + 256] + bta[tid + 256]);
}

// ----------------------------------------------------------------- WMMA GEMM
// C[M,N] = A[M,K] * Bt[N,K]^T; 128x128 block tile, 8 waves (4x2), 32x64/wave.
// A and Bt tiles are double-buffered in LDS via global_load_async_to_lds_b128.
#define EP_QKV 0
#define EP_PROJ 1
#define EP_FC1 2
#define EP_FC2 3

#define LDS_TILE_U16 5120          // 128 rows * 40 u16 stride
#define LDS_TILE_B   10240

template<int EP>
__global__ __launch_bounds__(256)
void gemm_kernel(const u16* __restrict__ A, const u16* __restrict__ Bt,
                 int Ktot, int Nout, const float* __restrict__ bias,
                 u16* __restrict__ ob0, u16* __restrict__ ob1, u16* __restrict__ ob2,
                 const float* __restrict__ resid, float* __restrict__ fout) {
  __shared__ u16 sA[2 * LDS_TILE_U16];   // [buf][m][k] stride 40
  __shared__ u16 sB[2 * LDS_TILE_U16];   // [buf][n][k] stride 40
  const int tid = threadIdx.x;
  const int m0 = blockIdx.y * 128;
  const int n0 = blockIdx.x * 128;
  const int lane = tid & 31, wave = tid >> 5;
  const int g = lane >> 4, lr = lane & 15;
  const int wm = (wave >> 1) * 32, wn = (wave & 1) * 64;

  v8f zero = {};
  v8f acc[2][4];
#pragma unroll
  for (int i = 0; i < 2; i++)
#pragma unroll
    for (int j = 0; j < 4; j++) acc[i][j] = zero;

  // async-copy geometry: each thread moves 32B of A and 32B of B per tile
  const int arow = tid >> 1;                    // 0..127 (row of A / row of Bt)
  const int aseg = (tid & 1) * 16;              // element offset within 32-elem row
  const u32 ldsRow = (u32)(arow * 80 + (tid & 1) * 32);   // byte offset in tile
  const u32 ldsAb = (u32)(uintptr_t)&sA[0];
  const u32 ldsBb = (u32)(uintptr_t)&sB[0];
  const size_t gAoff = ((size_t)(m0 + arow) * Ktot + aseg) * 2;   // bytes
  const size_t gBoff = ((size_t)(n0 + arow) * Ktot + aseg) * 2;   // bytes

  // prologue: tile 0 -> buffer 0
  {
    u32 la = ldsAb + ldsRow, lb = ldsBb + ldsRow;
    u32 ga = (u32)gAoff,     gb = (u32)gBoff;
    async_ld16(la, ga, A);  async_ld16(la + 16, ga + 16, A);
    async_ld16(lb, gb, Bt); async_ld16(lb + 16, gb + 16, Bt);
  }

  const int nk = Ktot >> 5;
  for (int t = 0; t < nk; t++) {
    if (t + 1 < nk) {
      // issue tile t+1 into the other buffer, then wait for tile t only
      u32 sel = (u32)((t + 1) & 1) * LDS_TILE_B;
      u32 la = ldsAb + sel + ldsRow, lb = ldsBb + sel + ldsRow;
      u32 ga = (u32)(gAoff + ((size_t)(t + 1) << 6));   // +32 elems = 64B
      u32 gb = (u32)(gBoff + ((size_t)(t + 1) << 6));
      async_ld16(la, ga, A);  async_ld16(la + 16, ga + 16, A);
      async_ld16(lb, gb, Bt); async_ld16(lb + 16, gb + 16, Bt);
      asm volatile("s_wait_asynccnt 0x4" ::: "memory");
    } else {
      asm volatile("s_wait_asynccnt 0x0" ::: "memory");
    }
    __syncthreads();   // tile t visible to all waves

    const int sb = (t & 1) * LDS_TILE_U16;
    FragU af[2], bfr[4];
#pragma unroll
    for (int mt = 0; mt < 2; mt++) {
      int row = wm + mt * 16 + lr;
#pragma unroll
      for (int p = 0; p < 8; p++) {
        int k0 = 16 * (p >> 2) + 2 * (p & 3) + 8 * g;   // ISA A-frag K order
        af[mt].u[p] = *(const u32*)&sA[sb + row * 40 + k0];
      }
    }
#pragma unroll
    for (int nt = 0; nt < 4; nt++) {
      int col = wn + nt * 16 + lr;
#pragma unroll
      for (int p = 0; p < 8; p++) {
        int k0 = 16 * (p >> 2) + 2 * (p & 3) + 8 * g;   // ISA A/B-frag K order (same as A)
        bfr[nt].u[p] = *(const u32*)&sB[sb + col * 40 + k0];
      }
    }
#pragma unroll
    for (int mt = 0; mt < 2; mt++)
#pragma unroll
      for (int nt = 0; nt < 4; nt++)
        acc[mt][nt] = __builtin_amdgcn_wmma_f32_16x16x32_bf16(
            false, af[mt].v, false, bfr[nt].v, (short)0, acc[mt][nt], false, false);
    __syncthreads();   // done reading buffer (t&1); it may be overwritten next iter
  }

  // ------------------------------- epilogue (C/D layout: m = r + 8g, n = lr)
#pragma unroll
  for (int mt = 0; mt < 2; mt++) {
#pragma unroll
    for (int r = 0; r < 8; r++) {
      const int gm = m0 + wm + mt * 16 + r + 8 * g;
      // per-row precompute (hoisted out of the nt loop)
      size_t rowbase = 0;
      if (EP == EP_QKV) {
        int win = gm / 49, t = gm % 49;
        rowbase = (size_t)win * 18816 + (size_t)t * 32;   // + nh*1568 + d later
      } else if (EP == EP_PROJ) {
        int b = gm / 3136, rem = gm % 3136;
        int wi = rem / 49, t = rem % 49;
        int h = (wi >> 3) * 7 + t / 7;
        int w = (wi & 7) * 7 + t % 7;
        int hf = h + 3; if (hf >= 56) hf -= 56;           // roll back (+3,+3)
        int wf = w + 3; if (wf >= 56) wf -= 56;
        rowbase = ((size_t)(b * 3136 + hf * 56 + wf)) * 384;
      } else if (EP == EP_FC1) {
        rowbase = (size_t)gm * 1536;
      } else {
        rowbase = (size_t)gm * 384;
      }
#pragma unroll
      for (int nt = 0; nt < 4; nt++) {
        int gn = n0 + wn + nt * 16 + lr;
        float val = acc[mt][nt][r] + bias[gn];
        if (EP == EP_QKV) {
          int s3 = gn / 384, c3 = gn % 384;
          int nh = c3 >> 5, d = c3 & 31;
          if (s3 == 0) val *= 0.17677669529663687f;       // HD^-0.5
          u16* dst = (s3 == 0) ? ob0 : (s3 == 1) ? ob1 : ob2;
          dst[rowbase + (size_t)nh * 1568 + d] = f2bf(val);
        } else if (EP == EP_PROJ) {
          size_t p = rowbase + gn;
          fout[p] = resid[p] + val;                       // shortcut + proj out
        } else if (EP == EP_FC1) {
          ob0[rowbase + gn] = f2bf(gelu_exact(val));
        } else { // EP_FC2
          size_t p = rowbase + gn;
          fout[p] = resid[p] + gelu_exact(val);           // x + gelu(fc2)
        }
      }
    }
  }
}

// --------------------------------------------------------- windowed attention
// One workgroup (4 waves) per (window, head). 49x49 padded to 64x64 in LDS.
__global__ __launch_bounds__(128)
void attn_kernel(const u16* __restrict__ qb, const u16* __restrict__ kb,
                 const u16* __restrict__ vb, const float* __restrict__ rel_bias,
                 u16* __restrict__ ao) {
  __shared__ u16 sQ[64 * 40];
  __shared__ u16 sK[64 * 40];
  __shared__ u16 sVt[32 * 72];   // transposed: [d][j]
  __shared__ u16 sP[64 * 72];    // softmax probs (bf16, A-matrix of 2nd WMMA)
  __shared__ float sS[64 * 68];  // fp32 scores
  const int head = blockIdx.x;
  const int win = head / 12, nh = head % 12;
  const int wl = win & 63, wrow = wl >> 3, wcol = wl & 7;
  const int tid = threadIdx.x;
  const u16* qg = qb + (size_t)head * 1568;   // 49*32
  const u16* kg = kb + (size_t)head * 1568;
  const u16* vg = vb + (size_t)head * 1568;
  for (int i = tid; i < 2048; i += 128) {     // 64x32 incl. zero pad rows
    int t = i >> 5, d = i & 31;
    u16 qv = (t < 49) ? qg[t * 32 + d] : (u16)0;
    u16 kv = (t < 49) ? kg[t * 32 + d] : (u16)0;
    u16 vv = (t < 49) ? vg[t * 32 + d] : (u16)0;
    sQ[t * 40 + d] = qv;
    sK[t * 40 + d] = kv;
    sVt[d * 72 + t] = vv;
  }
  __syncthreads();
  const int lane = tid & 31, wave = tid >> 5;
  const int g = lane >> 4, lr = lane & 15;

  // S = q * k^T (K=32, one wmma per 16x16 tile; wave owns one 16-row strip)
  {
    FragU a;
    int row = wave * 16 + lr;
#pragma unroll
    for (int p = 0; p < 8; p++) {
      int k0 = 16 * (p >> 2) + 2 * (p & 3) + 8 * g;
      a.u[p] = *(const u32*)&sQ[row * 40 + k0];
    }
#pragma unroll
    for (int nt = 0; nt < 4; nt++) {
      FragU b;
      int col = nt * 16 + lr;
#pragma unroll
      for (int p = 0; p < 8; p++) {
        int k0 = 16 * (p >> 2) + 2 * (p & 3) + 8 * g;
        b.u[p] = *(const u32*)&sK[col * 40 + k0];
      }
      v8f c = {};
      c = __builtin_amdgcn_wmma_f32_16x16x32_bf16(false, a.v, false, b.v,
                                                  (short)0, c, false, false);
#pragma unroll
      for (int r = 0; r < 8; r++) {
        int i2 = wave * 16 + r + 8 * g;
        int j2 = nt * 16 + lr;
        float sc;
        if (i2 < 49 && j2 < 49) {
          int ri = i2 / 7, ci = i2 % 7, rj = j2 / 7, cj = j2 % 7;
          sc = c[r] + rel_bias[((ri - rj + 6) * 13 + (ci - cj + 6)) * 12 + nh];
          int ghi = wrow * 7 + ri, gwi = wcol * 7 + ci;
          int ghj = wrow * 7 + rj, gwj = wcol * 7 + cj;
          int regi = (ghi < 49 ? 0 : (ghi < 53 ? 1 : 2)) * 3 + (gwi < 49 ? 0 : (gwi < 53 ? 1 : 2));
          int regj = (ghj < 49 ? 0 : (ghj < 53 ? 1 : 2)) * 3 + (gwj < 49 ? 0 : (gwj < 53 ? 1 : 2));
          if (regi != regj) sc -= 100.0f;
        } else {
          sc = -1e30f;
        }
        sS[i2 * 68 + j2] = sc;
      }
    }
  }
  __syncthreads();
  // row softmax over 49 valid cols (fp32), write normalized probs as bf16
  if (tid < 64) {
    int r = tid;
    if (r < 49) {
      float m = -1e30f;
      for (int j = 0; j < 49; j++) m = fmaxf(m, sS[r * 68 + j]);
      float sum = 0.f;
      for (int j = 0; j < 49; j++) {
        float ev = __expf(sS[r * 68 + j] - m);
        sS[r * 68 + j] = ev;
        sum += ev;
      }
      float inv = 1.0f / sum;
      for (int j = 0; j < 49; j++) sP[r * 72 + j] = f2bf(sS[r * 68 + j] * inv);
      for (int j = 49; j < 64; j++) sP[r * 72 + j] = 0;
    } else {
      for (int j = 0; j < 64; j++) sP[r * 72 + j] = 0;
    }
  }
  __syncthreads();
  // O = P * V  (M=64 padded, N=32, K=64 padded -> 2 k-steps)
  {
    v8f z = {};
    v8f c2[2]; c2[0] = z; c2[1] = z;
#pragma unroll
    for (int ks = 0; ks < 2; ks++) {
      FragU a;
      int row = wave * 16 + lr;
#pragma unroll
      for (int p = 0; p < 8; p++) {
        int k0 = 32 * ks + 16 * (p >> 2) + 2 * (p & 3) + 8 * g;
        a.u[p] = *(const u32*)&sP[row * 72 + k0];
      }
#pragma unroll
      for (int nt = 0; nt < 2; nt++) {
        FragU b;
        int col = nt * 16 + lr;
#pragma unroll
        for (int p = 0; p < 8; p++) {
          int j0 = 32 * ks + 16 * (p >> 2) + 2 * (p & 3) + 8 * g;
          b.u[p] = *(const u32*)&sVt[col * 72 + j0];
        }
        c2[nt] = __builtin_amdgcn_wmma_f32_16x16x32_bf16(
            false, a.v, false, b.v, (short)0, c2[nt], false, false);
      }
    }
#pragma unroll
    for (int nt = 0; nt < 2; nt++) {
#pragma unroll
      for (int r = 0; r < 8; r++) {
        int i2 = wave * 16 + r + 8 * g;
        int d = nt * 16 + lr;
        if (i2 < 49)
          ao[((size_t)win * 49 + i2) * 384 + nh * 32 + d] = f2bf(c2[nt][r]);
      }
    }
  }
}

// ------------------------------------------------------------------- launch
extern "C" void kernel_launch(void* const* d_in, const int* in_sizes, int n_in,
                              void* d_out, int out_size, void* d_ws, size_t ws_size,
                              hipStream_t stream) {
  const float* x      = (const float*)d_in[0];
  const float* ln1_g  = (const float*)d_in[1];
  const float* ln1_b  = (const float*)d_in[2];
  const float* qkv_w  = (const float*)d_in[3];
  const float* qkv_b  = (const float*)d_in[4];
  const float* rel_b  = (const float*)d_in[5];
  const float* proj_w = (const float*)d_in[6];
  const float* proj_b = (const float*)d_in[7];
  const float* ln2_g  = (const float*)d_in[8];
  const float* ln2_b  = (const float*)d_in[9];
  const float* fc1_w  = (const float*)d_in[10];
  const float* fc1_b  = (const float*)d_in[11];
  const float* fc2_w  = (const float*)d_in[12];
  const float* fc2_b  = (const float*)d_in[13];
  float* out = (float*)d_out;

  char* ws = (char*)d_ws;
  // bf16 weights, stored transposed [N][K] (front of workspace, ~3.4 MB)
  u16* wqkv  = (u16*)ws;            // [1152][384]
  u16* wproj = wqkv + 442368;       // [384][384]
  u16* wfc1  = wproj + 147456;      // [1536][384]
  u16* wfc2  = wfc1 + 589824;       // [384][1536]
  // activation buffers (lifetime-based reuse)
  const size_t OFF_A = 4ull << 20;                   // 4 MB
  const size_t SZ_BF = 100352ull * 384 * 2;          // 77,070,336 B
  u16*   bufA = (u16*)(ws + OFF_A);                  // win / attn_out / ln2 bf16
  u16*   qbuf = (u16*)(ws + OFF_A + 1 * SZ_BF);
  u16*   kbuf = (u16*)(ws + OFF_A + 2 * SZ_BF);
  u16*   vbuf = (u16*)(ws + OFF_A + 3 * SZ_BF);
  u16*   hbuf = qbuf;                                // FC1 out: 4*SZ_BF, reuses q/k/v
  float* xnew = (float*)(ws + OFF_A + 5 * SZ_BF);    // fp32 residual stream

  // 1) weights -> bf16, transposed to [N][K]
  cvt_bf16_t_kernel<<<(442368 + 255) / 256, 256, 0, stream>>>(qkv_w, wqkv, 384, 1152);
  cvt_bf16_t_kernel<<<(147456 + 255) / 256, 256, 0, stream>>>(proj_w, wproj, 384, 384);
  cvt_bf16_t_kernel<<<(589824 + 255) / 256, 256, 0, stream>>>(fc1_w, wfc1, 384, 1536);
  cvt_bf16_t_kernel<<<(589824 + 255) / 256, 256, 0, stream>>>(fc2_w, wfc2, 1536, 384);
  // 2) LN1 + roll(-3,-3) + window partition -> bf16 (window,token) rows
  ln_kernel<0><<<100352, 128, 0, stream>>>(x, ln1_g, ln1_b, bufA);
  // 3) QKV GEMM 100352x384x1152, scatter to per-head q/k/v (q pre-scaled)
  gemm_kernel<EP_QKV><<<dim3(9, 784), 256, 0, stream>>>(
      bufA, wqkv, 384, 1152, qkv_b, qbuf, kbuf, vbuf, nullptr, nullptr);
  // 4) attention: 2048 windows x 12 heads
  attn_kernel<<<24576, 128, 0, stream>>>(qbuf, kbuf, vbuf, rel_b, bufA);
  // 5) proj GEMM + window reverse + roll(+3,+3) + residual -> xnew (fp32)
  gemm_kernel<EP_PROJ><<<dim3(3, 784), 256, 0, stream>>>(
      bufA, wproj, 384, 384, proj_b, nullptr, nullptr, nullptr, x, xnew);
  // 6) LN2 -> bf16
  ln_kernel<1><<<100352, 128, 0, stream>>>(xnew, ln2_g, ln2_b, bufA);
  // 7) FC1 + GELU -> bf16
  gemm_kernel<EP_FC1><<<dim3(12, 784), 256, 0, stream>>>(
      bufA, wfc1, 384, 1536, fc1_b, hbuf, nullptr, nullptr, nullptr, nullptr);
  // 8) FC2 + GELU + residual -> d_out (fp32)
  gemm_kernel<EP_FC2><<<dim3(3, 784), 256, 0, stream>>>(
      hbuf, wfc2, 1536, 384, fc2_b, nullptr, nullptr, nullptr, xnew, out);
}
